// LlamaAttention_Tender_Eval_8117488190104
// MI455X (gfx1250) — compile-verified, matrix operand order checked by reading, not run on hardware
//
#include <hip/hip_runtime.h>
#include <math.h>

typedef _Float16 v16h __attribute__((ext_vector_type(16)));
typedef _Float16 v8h  __attribute__((ext_vector_type(8)));
typedef _Float16 v4h  __attribute__((ext_vector_type(4)));
typedef float    v8f  __attribute__((ext_vector_type(8)));
typedef int      v4i  __attribute__((vector_size(16)));

#define AS1 __attribute__((address_space(1)))
#define AS3 __attribute__((address_space(3)))

#if __has_builtin(__builtin_amdgcn_global_load_async_to_lds_b128)
#define HAS_ASYNC_LDS 1
#else
#define HAS_ASYNC_LDS 0
#endif

__device__ inline void wait_async0() {
#if __has_builtin(__builtin_amdgcn_s_wait_asynccnt)
  __builtin_amdgcn_s_wait_asynccnt(0);
#else
  asm volatile("s_wait_asynccnt 0x0" ::: "memory");
#endif
}

#define SEQ      2048
#define HIDDEN   4096
#define NQ_COLS  4096
#define NKV_COLS 1024
#define HEAD_DIM 128
#define N_HEADS  32
#define N_KVH    8

__device__ inline v8f v8f_zero() {
  v8f z;
#pragma unroll
  for (int i = 0; i < 8; ++i) z[i] = 0.0f;
  return z;
}

__device__ inline float half16_max(float x) {
  x = fmaxf(x, __shfl_xor(x, 1, 32));
  x = fmaxf(x, __shfl_xor(x, 2, 32));
  x = fmaxf(x, __shfl_xor(x, 4, 32));
  x = fmaxf(x, __shfl_xor(x, 8, 32));
  return x;
}
__device__ inline float half16_sum(float x) {
  x += __shfl_xor(x, 1, 32);
  x += __shfl_xor(x, 2, 32);
  x += __shfl_xor(x, 4, 32);
  x += __shfl_xor(x, 8, 32);
  return x;
}

// ---------------------------------------------------------------------------
// Tiled WMMA GEMM with double-buffered LDS: C[M,N] = X[M,K] @ W[N,K]^T
//   X: fp32 (XF16=false) or f16 (XF16=true); W: fp32 (converted at stash)
//   Out: f16 (OUTF32=false) or fp32 (OUTF32=true)
// Tile: 128x64 per 256-thread block, 8 waves, wave w -> rows [16w,16w+16)
// When X is already f16, its tile staging is a raw byte copy -> use
// GLOBAL_LOAD_ASYNC_TO_LDS_B128 (ASYNCcnt) when the toolchain exposes it.
// ---------------------------------------------------------------------------
#define TILE_M 128
#define TILE_N 64
#define TILE_K 32
#define LDS_STRIDE 40   // 32 + 8 halves pad; 80B row stride keeps 16B alignment

template<bool XF16, bool OUTF32>
__global__ __launch_bounds__(256)
void gemm_wmma(const void* __restrict__ Xv, const float* __restrict__ W,
               void* __restrict__ Outv, int M, int N, int K)
{
  __shared__ __align__(16) _Float16 sX[2][TILE_M][LDS_STRIDE];
  __shared__ __align__(16) _Float16 sW[2][TILE_N][LDS_STRIDE];

  const int tid  = threadIdx.x;
  const int wave = tid >> 5;      // 0..7
  const int lane = tid & 31;
  const int rowBase = blockIdx.x * TILE_M;
  const int colBase = blockIdx.y * TILE_N;

  const float*    Xf = (const float*)Xv;
  const _Float16* Xh = (const _Float16*)Xv;

  v8f acc[4];
#pragma unroll
  for (int i = 0; i < 4; ++i) acc[i] = v8f_zero();

  const int m0    = lane & 15;
  const int khalf = (lane >> 4) * 8;   // 0 or 8 (16-bit A/B fragment K pattern)
  const int cs = (tid & 7) * 4;        // staging column start
  const int r0 = tid >> 3;             // staging row base 0..31

  // register staging for the next tile (converted to f16 only at stash time)
  float4 rxf[4]; v4h rxh[4]; float4 rwf[2];

  auto fetchX = [&](int kk) {
#pragma unroll
    for (int i = 0; i < 4; ++i) {
      int r = r0 + i * 32;
      if constexpr (XF16)
        rxh[i] = *(const v4h*)(Xh + (size_t)(rowBase + r) * K + (kk + cs));
      else
        rxf[i] = *(const float4*)(Xf + (size_t)(rowBase + r) * K + (kk + cs));
    }
  };
  auto stashX = [&](int b) {
#pragma unroll
    for (int i = 0; i < 4; ++i) {
      int r = r0 + i * 32;
      if constexpr (XF16) {
        *(v4h*)&sX[b][r][cs] = rxh[i];
      } else {
        v4h h; h[0] = (_Float16)rxf[i].x; h[1] = (_Float16)rxf[i].y;
               h[2] = (_Float16)rxf[i].z; h[3] = (_Float16)rxf[i].w;
        *(v4h*)&sX[b][r][cs] = h;
      }
    }
  };
  auto fetchW = [&](int kk) {
#pragma unroll
    for (int i = 0; i < 2; ++i) {
      int r = r0 + i * 32;
      rwf[i] = *(const float4*)(W + (size_t)(colBase + r) * K + (kk + cs));
    }
  };
  auto stashW = [&](int b) {
#pragma unroll
    for (int i = 0; i < 2; ++i) {
      int r = r0 + i * 32;
      v4h h; h[0] = (_Float16)rwf[i].x; h[1] = (_Float16)rwf[i].y;
             h[2] = (_Float16)rwf[i].z; h[3] = (_Float16)rwf[i].w;
      *(v4h*)&sW[b][r][cs] = h;
    }
  };

#if HAS_ASYNC_LDS
  // Raw f16 tile copy: 128 rows x 64B = 512 x 16B chunks, 2 per thread.
  auto stageX_async = [&](int kk, int b) {
    if constexpr (XF16) {
#pragma unroll
      for (int i = 0; i < 2; ++i) {
        int c   = tid + i * 256;
        int row = c >> 2;
        int q   = (c & 3) * 8;                   // half-element offset in row
        const _Float16* g = Xh + (size_t)(rowBase + row) * K + kk + q;
        AS1 v4i* gp = (AS1 v4i*)(unsigned long long)(const void*)g;
        AS3 v4i* lp = (AS3 v4i*)(unsigned)(unsigned long long)&sX[b][row][q];
        __builtin_amdgcn_global_load_async_to_lds_b128(gp, lp, 0, 0);
      }
    }
  };
#endif

  int buf = 0;
#if HAS_ASYNC_LDS
  if constexpr (XF16) {
    stageX_async(0, 0);
    fetchW(0); stashW(0);
    wait_async0();
  } else {
    fetchX(0); fetchW(0); stashX(0); stashW(0);
  }
#else
  fetchX(0); fetchW(0); stashX(0); stashW(0);
#endif

  for (int k0 = 0; k0 < K; k0 += TILE_K) {
    const bool hasNext = (k0 + TILE_K) < K;

#if HAS_ASYNC_LDS
    if constexpr (XF16) {
      if (hasNext) fetchW(k0 + TILE_K);     // W regs: any time is safe
      __syncthreads();                      // current tile published
      if (hasNext) stageX_async(k0 + TILE_K, buf ^ 1); // prev readers done
    } else {
      if (hasNext) { fetchX(k0 + TILE_K); fetchW(k0 + TILE_K); }
      __syncthreads();
    }
#else
    if (hasNext) { fetchX(k0 + TILE_K); fetchW(k0 + TILE_K); }
    __syncthreads();
#endif

    v16h a;
    {
      int row = wave * 16 + m0;
      ((v8h*)&a)[0] = *(const v8h*)&sX[buf][row][khalf];
      ((v8h*)&a)[1] = *(const v8h*)&sX[buf][row][16 + khalf];
    }
#pragma unroll
    for (int nt = 0; nt < 4; ++nt) {
      v16h b;
      int col = nt * 16 + m0;
      ((v8h*)&b)[0] = *(const v8h*)&sW[buf][col][khalf];
      ((v8h*)&b)[1] = *(const v8h*)&sW[buf][col][16 + khalf];
      acc[nt] = __builtin_amdgcn_wmma_f32_16x16x32_f16(
          false, a, false, b, (short)0, acc[nt], false, false);
    }

#if HAS_ASYNC_LDS
    if constexpr (XF16) {
      if (hasNext) stashW(buf ^ 1);
      wait_async0();                        // ASYNCcnt->0 before next barrier
    } else {
      if (hasNext) { stashX(buf ^ 1); stashW(buf ^ 1); }
    }
#else
    if (hasNext) { stashX(buf ^ 1); stashW(buf ^ 1); }
#endif
    buf ^= 1;
  }

  // ---- epilogue: C layout = lane-half selects M+8, vgpr r = row, lane&15 = col ----
  const int rbase = rowBase + wave * 16 + ((lane >= 16) ? 8 : 0);
  const int c0    = colBase + (lane & 15);
#pragma unroll
  for (int nt = 0; nt < 4; ++nt) {
#pragma unroll
    for (int r = 0; r < 8; ++r) {
      size_t idx = (size_t)(rbase + r) * N + (c0 + nt * 16);
      if constexpr (OUTF32) ((float*)Outv)[idx] = acc[nt][r];
      else                  ((_Float16*)Outv)[idx] = (_Float16)acc[nt][r];
    }
  }
}

// ---------------------------------------------------------------------------
// RoPE + symmetric int8 quant/dequant.
// grid (SEQ, 48): y<32 -> Q head (rope), 32..39 -> K head (rope+quant),
// 40..47 -> V head (quant). 64 threads, thread t handles dims t and t+64.
// ---------------------------------------------------------------------------
__global__ __launch_bounds__(64)
void rope_quant(_Float16* __restrict__ Q, _Float16* __restrict__ K,
                _Float16* __restrict__ V, const int* __restrict__ pos)
{
  __shared__ float red[64];
  const int token = blockIdx.x;
  const int y = blockIdx.y;
  const int t = threadIdx.x;

  _Float16* row;
  bool doRope, doQuant;
  if (y < 32)      { row = Q + (size_t)token * NQ_COLS  + y * HEAD_DIM;        doRope = true;  doQuant = false; }
  else if (y < 40) { row = K + (size_t)token * NKV_COLS + (y - 32) * HEAD_DIM; doRope = true;  doQuant = true;  }
  else             { row = V + (size_t)token * NKV_COLS + (y - 40) * HEAD_DIM; doRope = false; doQuant = true;  }

  float x0 = (float)row[t];
  float x1 = (float)row[t + 64];

  if (doRope) {
    float p    = (float)pos[token];
    float invf = __powf(10000.0f, -(float)t * (1.0f / 64.0f));
    float ang  = p * invf;
    float c = __cosf(ang), s = __sinf(ang);
    float nx0 = x0 * c - x1 * s;       // rotate_half: first half pairs with -x2
    float nx1 = x1 * c + x0 * s;
    x0 = nx0; x1 = nx1;
  }
  if (doQuant) {
    red[t] = fmaxf(fabsf(x0), fabsf(x1));
    __syncthreads();
#pragma unroll
    for (int off = 32; off > 0; off >>= 1) {
      if (t < off) red[t] = fmaxf(red[t], red[t + off]);
      __syncthreads();
    }
    float scale = fmaxf(red[0] * (1.0f / 127.0f), 1e-9f);
    x0 = fminf(fmaxf(rintf(x0 / scale), -128.0f), 127.0f) * scale;
    x1 = fminf(fmaxf(rintf(x1 / scale), -128.0f), 127.0f) * scale;
  }
  row[t]      = (_Float16)x0;
  row[t + 64] = (_Float16)x1;
}

// ---------------------------------------------------------------------------
// Flash attention, causal, GQA (4 Q heads per KV head).
// 128 threads = 4 waves per block; block covers 64 query rows of one head.
// ---------------------------------------------------------------------------
__global__ __launch_bounds__(128)
void attn_flash(const _Float16* __restrict__ Q, const _Float16* __restrict__ K,
                const _Float16* __restrict__ V, _Float16* __restrict__ O)
{
  __shared__ float sP[4][16][33];
  __shared__ __align__(16) _Float16 sVt[128][LDS_STRIDE]; // V^T: [col][key]

  const int tid   = threadIdx.x;
  const int wave  = tid >> 5;
  const int lane  = tid & 31;
  const int h     = blockIdx.y;
  const int kvh   = h >> 2;                // N_HEADS / N_KVH = 4
  const int m0    = lane & 15;
  const int hi    = lane >> 4;             // lane half
  const int khalf = hi * 8;
  const int qrow0 = blockIdx.x * 64 + wave * 16;
  const int qlast = qrow0 + 15;
  const float sm_scale = 0.08838834764831845f; // 1/sqrt(128)

  // Q fragments: 4 x (16x32) over feature dim
  v16h qa[4];
  {
    const _Float16* qrow = Q + (size_t)(qrow0 + m0) * NQ_COLS + h * HEAD_DIM;
#pragma unroll
    for (int f = 0; f < 4; ++f) {
      ((v8h*)&qa[f])[0] = *(const v8h*)(qrow + f * 32 + khalf);
      ((v8h*)&qa[f])[1] = *(const v8h*)(qrow + f * 32 + 16 + khalf);
    }
  }

  v8f o[8];
#pragma unroll
  for (int nt = 0; nt < 8; ++nt) o[nt] = v8f_zero();
  float mrun[8], lrun[8];
#pragma unroll
  for (int r = 0; r < 8; ++r) { mrun[r] = -1e30f; lrun[r] = 0.0f; }

  const int nkb = (blockIdx.x * 64 + 64 + 31) >> 5;  // 32-key chunks for block

  for (int kc = 0; kc < nkb; ++kc) {
    const int K0 = kc * 32;
    __syncthreads();   // all prior-iteration LDS readers done before restaging

    // ---- stage V chunk transposed (cooperative, 512 v8h chunks / 128 thr) ----
#pragma unroll
    for (int it = 0; it < 4; ++it) {
      int c8 = tid + it * 128;
      int vr = c8 >> 4;          // key 0..31
      int vc = (c8 & 15) * 8;    // col
      v8h vv = *(const v8h*)(V + (size_t)(K0 + vr) * NKV_COLS + kvh * HEAD_DIM + vc);
#pragma unroll
      for (int j = 0; j < 8; ++j) sVt[vc + j][vr] = vv[j];
    }

    const bool active = (K0 <= qlast);
    if (active) {
      // ---- load ALL K B-fragments first so loads clause & pipeline ----
      v16h b0[4], b1[4];
#pragma unroll
      for (int f = 0; f < 4; ++f) {
        const _Float16* kr0 = K + (size_t)(K0 + m0) * NKV_COLS      + kvh * HEAD_DIM + f * 32;
        const _Float16* kr1 = K + (size_t)(K0 + 16 + m0) * NKV_COLS + kvh * HEAD_DIM + f * 32;
        ((v8h*)&b0[f])[0] = *(const v8h*)(kr0 + khalf);
        ((v8h*)&b0[f])[1] = *(const v8h*)(kr0 + 16 + khalf);
        ((v8h*)&b1[f])[0] = *(const v8h*)(kr1 + khalf);
        ((v8h*)&b1[f])[1] = *(const v8h*)(kr1 + 16 + khalf);
      }
      v8f s0 = v8f_zero(), s1 = v8f_zero();
#pragma unroll
      for (int f = 0; f < 4; ++f)
        s0 = __builtin_amdgcn_wmma_f32_16x16x32_f16(false, qa[f], false, b0[f], (short)0, s0, false, false);
#pragma unroll
      for (int f = 0; f < 4; ++f)
        s1 = __builtin_amdgcn_wmma_f32_16x16x32_f16(false, qa[f], false, b1[f], (short)0, s1, false, false);

      // ---- causal mask + online softmax (rows per lane-half + vgpr) ----
      const int j0 = K0 + m0;
      const int j1 = K0 + 16 + m0;
      float alpha[8];
#pragma unroll
      for (int r = 0; r < 8; ++r) {
        int i = qrow0 + r + hi * 8;
        float v0 = s0[r] * sm_scale; if (j0 > i) v0 = -1e9f;
        float v1 = s1[r] * sm_scale; if (j1 > i) v1 = -1e9f;
        float mx = half16_max(fmaxf(v0, v1));
        float mn = fmaxf(mrun[r], mx);
        alpha[r] = __expf(mrun[r] - mn);
        float p0 = __expf(v0 - mn);
        float p1 = __expf(v1 - mn);
        s0[r] = p0; s1[r] = p1;
        lrun[r] = lrun[r] * alpha[r] + half16_sum(p0 + p1);
        mrun[r] = mn;
      }
#pragma unroll
      for (int nt = 0; nt < 8; ++nt)
#pragma unroll
        for (int r = 0; r < 8; ++r) o[nt][r] *= alpha[r];

      // ---- P: C-layout -> A-fragment layout via per-wave LDS region ----
#pragma unroll
      for (int r = 0; r < 8; ++r) {
        int m = r + hi * 8;
        sP[wave][m][m0]      = s0[r];
        sP[wave][m][16 + m0] = s1[r];
      }
    }
    __syncthreads();   // sVt + sP writes visible to all lanes

    if (active) {
      v16h pa;
#pragma unroll
      for (int j = 0; j < 8; ++j) {
        ((_Float16*)&pa)[j]     = (_Float16)sP[wave][m0][khalf + j];
        ((_Float16*)&pa)[8 + j] = (_Float16)sP[wave][m0][16 + khalf + j];
      }
      // ---- PV: 8 WMMAs over 128 output cols, B from transposed V in LDS ----
#pragma unroll
      for (int nt = 0; nt < 8; ++nt) {
        v16h bv;
        const _Float16* vcol = &sVt[nt * 16 + m0][0];
        ((v8h*)&bv)[0] = *(const v8h*)(vcol + khalf);
        ((v8h*)&bv)[1] = *(const v8h*)(vcol + 16 + khalf);
        o[nt] = __builtin_amdgcn_wmma_f32_16x16x32_f16(false, pa, false, bv, (short)0, o[nt], false, false);
      }
    }
  }

  // ---- normalize + write attention output (f16, [seq][heads*128]) ----
#pragma unroll
  for (int nt = 0; nt < 8; ++nt) {
#pragma unroll
    for (int r = 0; r < 8; ++r) {
      int row = qrow0 + r + hi * 8;
      int col = h * HEAD_DIM + nt * 16 + m0;
      O[(size_t)row * NQ_COLS + col] = (_Float16)(o[nt][r] / lrun[r]);
    }
  }
}

// ---------------------------------------------------------------------------
extern "C" void kernel_launch(void* const* d_in, const int* in_sizes, int n_in,
                              void* d_out, int out_size, void* d_ws, size_t ws_size,
                              hipStream_t stream) {
  const float* hs = (const float*)d_in[0];   // [1,2048,4096] fp32
  const float* wq = (const float*)d_in[1];   // [4096,4096]
  const float* wk = (const float*)d_in[2];   // [1024,4096]
  const float* wv = (const float*)d_in[3];   // [1024,4096]
  const float* wo = (const float*)d_in[4];   // [4096,4096]
  const int*  pos = (const int*)d_in[5];     // [1,2048] int32

  char* ws = (char*)d_ws;
  _Float16* Qw = (_Float16*)ws;                                           // 16 MiB
  _Float16* Kw = (_Float16*)(ws + (size_t)SEQ * NQ_COLS * 2);             //  4 MiB
  _Float16* Vw = (_Float16*)(ws + (size_t)SEQ * (NQ_COLS + NKV_COLS) * 2);//  4 MiB
  _Float16* Aw = (_Float16*)(ws + (size_t)SEQ * (NQ_COLS + 2 * NKV_COLS) * 2); // 16 MiB

  dim3 blk(256);
  // QKV projections (fp32 in -> f16 out, fp32 weights converted in LDS stage)
  gemm_wmma<false, false><<<dim3(SEQ / TILE_M, NQ_COLS / TILE_N),  blk, 0, stream>>>(hs, wq, Qw, SEQ, NQ_COLS,  HIDDEN);
  gemm_wmma<false, false><<<dim3(SEQ / TILE_M, NKV_COLS / TILE_N), blk, 0, stream>>>(hs, wk, Kw, SEQ, NKV_COLS, HIDDEN);
  gemm_wmma<false, false><<<dim3(SEQ / TILE_M, NKV_COLS / TILE_N), blk, 0, stream>>>(hs, wv, Vw, SEQ, NKV_COLS, HIDDEN);
  // RoPE (Q,K) + int8 quant-dequant (K,V)
  rope_quant<<<dim3(SEQ, 48), dim3(64), 0, stream>>>(Qw, Kw, Vw, pos);
  // causal GQA flash attention (4 waves / 64 query rows per block)
  attn_flash<<<dim3(SEQ / 64, N_HEADS), dim3(128), 0, stream>>>(Qw, Kw, Vw, Aw);
  // output projection (f16 in -> fp32 out; async LDS staging when available)
  gemm_wmma<true, true><<<dim3(SEQ / TILE_M, NQ_COLS / TILE_N), blk, 0, stream>>>(Aw, wo, (float*)d_out, SEQ, NQ_COLS, NQ_COLS);
}
